// WCT_35304631173366
// MI455X (gfx1250) — matrix operand order repaired and checked by reading, not compiled, not run on hardware
//
#include <hip/hip_runtime.h>
#include <hip/hip_bf16.h>

// ---------------------------------------------------------------------------
// WCT (whitening-coloring transform) for MI455X / gfx1250.
// wave32 + v_wmma_f32_16x16x32_bf16 + Tensor Data Mover tile staging.
// B=4, C=64, N=512*512. Gram (64xNx64) and apply (64x64xN) GEMMs run as bf16
// WMMA with f32 accumulation; tiles staged global->LDS by TDM (pipelined:
// DMA of tile t+1 overlaps WMMA of tile t), converted once to packed-bf16 LDS,
// operands then load as contiguous ds_load_b128 straight into WMMA registers.
// ---------------------------------------------------------------------------

#define B_      4
#define C_      64
#define N_      262144      // 512*512
#define TILE_N  128
#define GTILES  16          // tiles per gram workgroup (chunk = 2048 cols)

typedef __attribute__((ext_vector_type(16))) __bf16   v16bf;
typedef __attribute__((ext_vector_type(8)))  __bf16   v8bf;
typedef __attribute__((ext_vector_type(8)))  float    v8f;
typedef __attribute__((ext_vector_type(8)))  unsigned v8u;
typedef __attribute__((ext_vector_type(4)))  unsigned u32x4;
typedef __attribute__((ext_vector_type(8)))  int      i32x8;
typedef __attribute__((ext_vector_type(4)))  int      i32x4;

// pack two f32 into two truncated bf16 in ONE v_perm_b32
static __device__ __forceinline__ unsigned pack2(float x, float y) {
    unsigned ux = __builtin_bit_cast(unsigned, x);
    unsigned uy = __builtin_bit_cast(unsigned, y);
    // result bytes {0,1}=x[2],x[3]; {2,3}=y[2],y[3]
    return __builtin_amdgcn_perm(uy, ux, 0x07060302u);
}
// concat two 16-byte LDS chunks into a v16bf operand (no repacking movs)
static __device__ __forceinline__ v16bf ld_op16(const void* p0, const void* p1) {
    v8bf lo = __builtin_bit_cast(v8bf, *(const uint4*)p0);
    v8bf hi = __builtin_bit_cast(v8bf, *(const uint4*)p1);
    return __builtin_shufflevector(lo, hi, 0, 1, 2, 3, 4, 5, 6, 7,
                                           8, 9, 10, 11, 12, 13, 14, 15);
}
// generic pointer to LDS: low 32 bits are the LDS byte offset (flat aperture)
static __device__ __forceinline__ unsigned lds_addr32(const void* p) {
    return (unsigned)(unsigned long long)(uintptr_t)p;
}

// ---------------------------------------------------------------------------
// TDM: DMA a [tile_d1 x tile_d0] f32 tile (row stride = N_ elements) into LDS.
// D# packed per CDNA5 ISA 8.3/8.4 (2D tensor, groups 2/3 zero, count=1).
// 6-arg builtin on this toolchain. EXEC ignored; issue from one wave only.
// ---------------------------------------------------------------------------
static __device__ __forceinline__ void tdm_load_tile_f32(unsigned lds_off,
                                                         const float* gptr,
                                                         unsigned tile_d0,
                                                         unsigned tile_d1) {
    unsigned long long ga = (unsigned long long)(uintptr_t)gptr;
    u32x4 g0;
    g0[0] = 1u;                                            // count=1 (valid user D#)
    g0[1] = lds_off;                                       // lds_addr (bytes)
    g0[2] = (unsigned)(ga & 0xffffffffull);                // global_addr[31:0]
    g0[3] = (unsigned)((ga >> 32) & 0x1ffffffull)          // global_addr[56:32]
          | (2u << 30);                                    // type=2 ("image")
    i32x8 g1;
    g1[0] = (int)(2u << 16);                               // data_size=2 -> 4 bytes
    g1[1] = (int)((N_ & 0xffffu) << 16);                   // tensor_dim0[15:0]
    g1[2] = (int)(((unsigned)N_ >> 16) | (64u << 16));     // tdim0[31:16] | tdim1[15:0]
    g1[3] = (int)(tile_d0 << 16);                          // tdim1[31:16]=0 | tile_dim0
    g1[4] = (int)tile_d1;                                  // tile_dim1 | tile_dim2=0
    g1[5] = (int)(unsigned)N_;                             // tensor_dim0_stride[31:0]
    g1[6] = 0;                                             // stride0[47:32] | stride1 lo
    g1[7] = 0;                                             // stride1 hi
    i32x4 z4 = {0, 0, 0, 0};
    i32x8 z8 = {0, 0, 0, 0, 0, 0, 0, 0};
    __builtin_amdgcn_tensor_load_to_lds(g0, g1, z4, z4, z8, 0);
}

// ---------------------------------------------------------------------------
// Kernel 1: per-(b,c) row sums of cF and sF (for means).
// ---------------------------------------------------------------------------
__global__ __launch_bounds__(256) void wct_means(const float* __restrict__ cF,
                                                 const float* __restrict__ sF,
                                                 float* __restrict__ csum,
                                                 float* __restrict__ ssum) {
    __shared__ float red[256];
    const int bc  = blockIdx.x;
    const int tid = threadIdx.x;
    const float4* pc = (const float4*)(cF + (long long)bc * N_);
    const float4* ps = (const float4*)(sF + (long long)bc * N_);
    float sc = 0.f, ss = 0.f;
    for (int j = tid; j < N_ / 4; j += 256) {
        float4 v = pc[j]; sc += (v.x + v.y) + (v.z + v.w);
        float4 u = ps[j]; ss += (u.x + u.y) + (u.z + u.w);
    }
    red[tid] = sc; __syncthreads();
    for (int s = 128; s > 0; s >>= 1) { if (tid < s) red[tid] += red[tid + s]; __syncthreads(); }
    if (tid == 0) csum[bc] = red[0];
    __syncthreads();
    red[tid] = ss; __syncthreads();
    for (int s = 128; s > 0; s >>= 1) { if (tid < s) red[tid] += red[tid + s]; __syncthreads(); }
    if (tid == 0) ssum[bc] = red[0];
}

// ---------------------------------------------------------------------------
// Kernel 2: uncentered Gram G[b] = cf[b]*cf[b]^T. grid (128 chunks, B).
// Software pipeline: TDM(t+1) -> f32buf overlaps pack-consume of tile t
// (WMMA reads only bfbuf, so no LDS conflict; wait moved to loop top).
// ---------------------------------------------------------------------------
__global__ __launch_bounds__(256) void wct_gram(const float* __restrict__ cF,
                                                float* __restrict__ G) {
    __shared__ float    f32buf[C_ * TILE_N];   // 32 KB staging (TDM target)
    __shared__ unsigned bfbuf[C_ * 68];        // packed bf16, 272B row pitch

    const int b    = blockIdx.y;
    const int tid  = threadIdx.x;
    const int lane = tid & 31;
    const int wave = tid >> 5;
    const int hi   = lane >> 4;
    const int l15  = lane & 15;
    const int ci   = wave >> 1;                // output block row (0..3)
    const int cj0  = (wave & 1) * 2;           // first output block col
    const long long nbase = (long long)blockIdx.x * (GTILES * TILE_N);
    const float* cfb = cF + (long long)b * C_ * N_;
    const unsigned ldsOff = lds_addr32(f32buf);
    const char* bfBase = (const char*)bfbuf;

    v8f acc0 = {};
    v8f acc1 = {};

    if (wave == 0)                              // prologue: DMA tile 0
        tdm_load_tile_f32(ldsOff, cfb + nbase, TILE_N, C_);

    for (int t = 0; t < GTILES; ++t) {
        if (wave == 0) __builtin_amdgcn_s_wait_tensorcnt(0);
        __syncthreads();   // tile t resident in f32buf; bfbuf free (prev consumed)

        // pack f32 tile -> bf16 tile: 64 rows x 64 dword-pairs = 4096 slots
#pragma unroll
        for (int s = 0; s < 16; ++s) {
            int slot = tid + s * 256;
            int row = slot >> 6, w = slot & 63;
            float2 f = *(const float2*)&f32buf[row * TILE_N + 2 * w];
            bfbuf[row * 68 + w] = pack2(f.x, f.y);
        }
        __syncthreads();   // bfbuf ready; f32buf free -> overlap next DMA

        if (wave == 0 && t + 1 < GTILES)
            tdm_load_tile_f32(ldsOff, cfb + nbase + (long long)(t + 1) * TILE_N,
                              TILE_N, C_);

#pragma unroll
        for (int ks = 0; ks < 4; ++ks) {
            // A: rows ci*16+l15; chunks K=e+16*hi             (16B each)
            const char* pa = bfBase + (ci * 16 + l15) * 272 + ks * 64 + hi * 32;
            v16bf a = ld_op16(pa, pa + 16);
            // B: rows cj*16+l15; chunks K=e+16*hi             (16B each)
            const char* pb0 = bfBase + (cj0 * 16 + l15) * 272 + ks * 64 + hi * 32;
            const char* pb1 = bfBase + ((cj0 + 1) * 16 + l15) * 272 + ks * 64 + hi * 32;
            v16bf b0 = ld_op16(pb0, pb0 + 16);
            v16bf b1 = ld_op16(pb1, pb1 + 16);
            acc0 = __builtin_amdgcn_wmma_f32_16x16x32_bf16(false, a, false, b0,
                                                           (short)0, acc0, false, false);
            acc1 = __builtin_amdgcn_wmma_f32_16x16x32_bf16(false, a, false, b1,
                                                           (short)0, acc1, false, false);
        }
        // no trailing barrier: next-iteration top barrier protects bfbuf
    }
#pragma unroll
    for (int r = 0; r < 8; ++r) {
        int M = r + 8 * hi;
        atomicAdd(&G[b * 4096 + (ci * 16 + M) * 64 + cj0 * 16 + l15],       acc0[r]);
        atomicAdd(&G[b * 4096 + (ci * 16 + M) * 64 + (cj0 + 1) * 16 + l15], acc1[r]);
    }
}

// ---------------------------------------------------------------------------
// Kernel 3: per-batch 64x64 solve (cov build, parallel Jacobi, cov^-1/2,
// coloring, fused T/bias). One 64-thread block per batch.
// ---------------------------------------------------------------------------
__global__ __launch_bounds__(64) void wct_solve(const float* __restrict__ G,
                                                const float* __restrict__ csum,
                                                const float* __restrict__ ssum,
                                                const float* __restrict__ alpha,
                                                const float* __restrict__ s_e,
                                                const float* __restrict__ s_v,
                                                float* __restrict__ Tm,
                                                float* __restrict__ biasg) {
    __shared__ float A[64 * 65];
    __shared__ float V[64 * 65];
    __shared__ float S2[64 * 65];
    __shared__ float ed[64];
    __shared__ int   pos[64];
    __shared__ float cs[32], sn[32];
    __shared__ int   pp[32], qq[32];

    const int b = blockIdx.x;
    const int t = threadIdx.x;
    const float invN = 1.0f / (float)N_;
    const float inv_nm1 = 1.0f / (float)(N_ - 1);

    const float cm_t = csum[b * 64 + t] * invN;
    for (int j = 0; j < 64; ++j) {
        float cmj = csum[b * 64 + j] * invN;
        float cov = (G[b * 4096 + t * 64 + j] - (float)N_ * cm_t * cmj) * inv_nm1
                    + ((t == j) ? 1.0f : 0.0f);
        A[t * 65 + j] = cov;
        V[t * 65 + j] = (t == j) ? 1.0f : 0.0f;
    }
    pos[t] = t;
    __syncthreads();

    for (int sweep = 0; sweep < 12; ++sweep) {
        for (int round = 0; round < 63; ++round) {
            if (t < 32) {
                int p = pos[t], q = pos[63 - t];
                if (p > q) { int tmp = p; p = q; q = tmp; }
                float app = A[p * 65 + p], aqq = A[q * 65 + q], apq = A[p * 65 + q];
                float c = 1.0f, s = 0.0f;
                if (fabsf(apq) > 1e-12f) {
                    float tau = (aqq - app) / (2.0f * apq);
                    float tt  = (tau >= 0.f ? 1.0f : -1.0f)
                              / (fabsf(tau) + sqrtf(1.0f + tau * tau));
                    c = rsqrtf(1.0f + tt * tt);
                    s = tt * c;
                }
                cs[t] = c; sn[t] = s; pp[t] = p; qq[t] = q;
            }
            __syncthreads();
            for (int k = 0; k < 32; ++k) {      // column rotations (row t)
                int p = pp[k], q = qq[k]; float c = cs[k], s = sn[k];
                float ap = A[t * 65 + p], aq = A[t * 65 + q];
                A[t * 65 + p] = c * ap - s * aq;
                A[t * 65 + q] = s * ap + c * aq;
                float vp = V[t * 65 + p], vq = V[t * 65 + q];
                V[t * 65 + p] = c * vp - s * vq;
                V[t * 65 + q] = s * vp + c * vq;
            }
            __syncthreads();
            for (int k = 0; k < 32; ++k) {      // row rotations (column t)
                int p = pp[k], q = qq[k]; float c = cs[k], s = sn[k];
                float ap = A[p * 65 + t], aq = A[q * 65 + t];
                A[p * 65 + t] = c * ap - s * aq;
                A[q * 65 + t] = s * ap + c * aq;
            }
            __syncthreads();
            if (t == 0) {                       // rotate tournament schedule
                int last = pos[63];
                for (int i = 63; i >= 2; --i) pos[i] = pos[i - 1];
                pos[1] = last;
            }
            __syncthreads();
        }
    }

    ed[t] = rsqrtf(fmaxf(A[t * 65 + t], 1e-20f));   // eig^-1/2 (eigs >= 1)
    __syncthreads();
    for (int l = 0; l < 64; ++l) {                   // S2 = V diag(ed) V^T
        float s = 0.f;
        for (int j = 0; j < 64; ++j) s += V[t * 65 + j] * ed[j] * V[l * 65 + j];
        S2[t * 65 + l] = s;
    }
    __syncthreads();
    ed[t] = sqrtf(fmaxf(s_e[b * 64 + t], 0.f));      // style eig^+1/2
    __syncthreads();
    const float* svb = s_v + b * 4096;               // colorM into A
    for (int l = 0; l < 64; ++l) {
        float s = 0.f;
        for (int j = 0; j < 64; ++j) s += svb[t * 64 + j] * ed[j] * svb[l * 64 + j];
        A[t * 65 + l] = s;
    }
    __syncthreads();
    for (int l = 0; l < 64; ++l) {                   // M = colorM * S2 into V
        float s = 0.f;
        for (int j = 0; j < 64; ++j) s += A[t * 65 + j] * S2[j * 65 + l];
        V[t * 65 + l] = s;
    }
    __syncthreads();
    const float al = alpha[0];
    float bsum = 0.f;
    for (int l = 0; l < 64; ++l) {
        float m = V[t * 65 + l];
        Tm[b * 4096 + t * 64 + l] = al * m + ((t == l) ? (1.0f - al) : 0.0f);
        bsum += m * (csum[b * 64 + l] * invN);
    }
    biasg[b * 64 + t] = al * (ssum[b * 64 + t] * invN - bsum);
}

// ---------------------------------------------------------------------------
// Kernel 4: out[b] = T[b]*cf[b] + bias[b]. grid (N/128, B), 8 waves/block.
// TDM stages 64x128 f32 tile; packed-transposed bf16 LDS [n][cpair] (pitch 36
// words = 144B rows) makes B operands contiguous b128 loads. A-operand build
// from the tiny T matrix overlaps the DMA.
// ---------------------------------------------------------------------------
__global__ __launch_bounds__(256) void wct_apply(const float* __restrict__ cF,
                                                 const float* __restrict__ Tm,
                                                 const float* __restrict__ biasg,
                                                 float* __restrict__ out) {
    __shared__ float    f32buf[C_ * TILE_N];   // 32 KB staging (TDM target)
    __shared__ unsigned tileT[TILE_N * 36];    // transposed packed bf16

    const int b    = blockIdx.y;
    const int tid  = threadIdx.x;
    const int lane = tid & 31;
    const int wave = tid >> 5;
    const int hi   = lane >> 4;
    const int l15  = lane & 15;
    const long long n0 = (long long)blockIdx.x * TILE_N;
    const float* cfb = cF + (long long)b * C_ * N_;
    const float* Tb  = Tm + b * 4096;

    if (wave == 0)
        tdm_load_tile_f32(lds_addr32(f32buf), cfb + n0, TILE_N, C_);

    // A operands from T (tiny, L2-resident) -- overlaps the TDM DMA
    v16bf Aop[4][2];
#pragma unroll
    for (int m = 0; m < 4; ++m)
#pragma unroll
        for (int kk = 0; kk < 2; ++kk) {
            v8u w;
#pragma unroll
            for (int v = 0; v < 8; ++v) {
                int K = 2 * v + 16 * hi + 32 * kk;
                float2 f = *(const float2*)&Tb[(m * 16 + l15) * 64 + K];
                w[v] = pack2(f.x, f.y);
            }
            Aop[m][kk] = __builtin_bit_cast(v16bf, w);
        }

    if (wave == 0) __builtin_amdgcn_s_wait_tensorcnt(0);
    __syncthreads();

    // transpose-pack: slots = 128 n x 32 channel-pairs
#pragma unroll
    for (int s = 0; s < 16; ++s) {
        int slot = tid + s * 256;
        int n = slot & 127, cp = slot >> 7;
        float x = f32buf[(2 * cp) * TILE_N + n];
        float y = f32buf[(2 * cp + 1) * TILE_N + n];
        tileT[n * 36 + cp] = pack2(x, y);
    }
    __syncthreads();

    const int colbase = wave * 16;
    const char* cb = (const char*)tileT + (colbase + l15) * 144;
    v16bf Bop0 = ld_op16(cb + 32 * hi,      cb + 32 * hi + 16);      // K 0..31
    v16bf Bop1 = ld_op16(cb + 64 + 32 * hi, cb + 64 + 32 * hi + 16); // K 32..63

#pragma unroll
    for (int m = 0; m < 4; ++m) {
        v8f acc;
#pragma unroll
        for (int r = 0; r < 8; ++r)
            acc[r] = biasg[b * 64 + m * 16 + r + 8 * hi];
        acc = __builtin_amdgcn_wmma_f32_16x16x32_bf16(false, Aop[m][0], false, Bop0,
                                                      (short)0, acc, false, false);
        acc = __builtin_amdgcn_wmma_f32_16x16x32_bf16(false, Aop[m][1], false, Bop1,
                                                      (short)0, acc, false, false);
#pragma unroll
        for (int r = 0; r < 8; ++r) {
            int c = m * 16 + r + 8 * hi;
            out[(long long)(b * 64 + c) * N_ + n0 + colbase + l15] = acc[r];
        }
    }
}

// ---------------------------------------------------------------------------
extern "C" void kernel_launch(void* const* d_in, const int* in_sizes, int n_in,
                              void* d_out, int out_size, void* d_ws, size_t ws_size,
                              hipStream_t stream) {
    const float* cF    = (const float*)d_in[0];
    const float* sF    = (const float*)d_in[1];
    const float* alpha = (const float*)d_in[2];
    const float* s_e   = (const float*)d_in[3];
    const float* s_v   = (const float*)d_in[4];
    float* out = (float*)d_out;

    float* G    = (float*)d_ws;   // 4*64*64 = 16384 floats (zeroed each call)
    float* csum = G + 16384;      // 256
    float* ssum = csum + 256;     // 256
    float* Tm   = ssum + 256;     // 16384
    float* bias = Tm + 16384;     // 256

    (void)hipMemsetAsync(G, 0, 16384 * sizeof(float), stream);
    wct_means<<<dim3(B_ * C_), dim3(256), 0, stream>>>(cF, sF, csum, ssum);
    wct_gram <<<dim3(N_ / (GTILES * TILE_N), B_), dim3(256), 0, stream>>>(cF, G);
    wct_solve<<<dim3(B_), dim3(64), 0, stream>>>(G, csum, ssum, alpha,
                                                 s_e, s_v, Tm, bias);
    wct_apply<<<dim3(N_ / TILE_N, B_), dim3(256), 0, stream>>>(cF, Tm, bias, out);
}
